// BinaryDiff_43894565765464
// MI455X (gfx1250) — matrix operand order repaired and checked by reading, not compiled, not run on hardware
//
#include <hip/hip_runtime.h>
#include <hip/hip_bf16.h>

// Problem: out[M,N] = x[M,K] @ (base + coeff*(2*mask-1))[K,N]
// M = B*S = 8192, K = 4096, N = 4096.  Single fused bf16 WMMA GEMM.
#define DIM_M 8192
#define DIM_K 4096
#define DIM_N 4096

typedef __attribute__((ext_vector_type(16))) __bf16 v16bf;
typedef __attribute__((ext_vector_type(8)))  __bf16 v8bf;
typedef __attribute__((ext_vector_type(8)))  float  v8f;
typedef __attribute__((ext_vector_type(4)))  int    v4i;

// Address-space-qualified int4 pointees for the async-to-LDS builtin
// (param types per hipcc diagnostic: int4* in global AS, int4* in LDS AS).
typedef __attribute__((address_space(1))) v4i v4i_g;
typedef __attribute__((address_space(3))) v4i v4i_l;

// gfx1250 async global->LDS path (ASYNCcnt), if the toolchain exposes it.
#if defined(__has_builtin)
#if __has_builtin(__builtin_amdgcn_global_load_async_to_lds_b128)
#define HAVE_ASYNC_LDS 1
#endif
#endif
#ifndef HAVE_ASYNC_LDS
#define HAVE_ASYNC_LDS 0
#endif

__device__ __forceinline__ unsigned short f32_to_bf16(float f) {
  unsigned int u = __builtin_bit_cast(unsigned int, f);
  u += 0x7FFFu + ((u >> 16) & 1u);           // round-to-nearest-even
  return (unsigned short)(u >> 16);
}

// ---------------------------------------------------------------------------
// Prep 1: W_eff = base + coeff*(2*mask-1), stored TRANSPOSED as WT[N][K] bf16
// so the GEMM B-tile loads are contiguous along K. LDS-tiled 64x64 transpose.
// ---------------------------------------------------------------------------
__global__ __launch_bounds__(256) void bd_prep_w(const float* __restrict__ base,
                                                 const int*   __restrict__ mask,
                                                 const float* __restrict__ coeff_p,
                                                 unsigned short* __restrict__ wt) {
  __shared__ float tile[64 * 65];            // +1 pad: conflict-free transpose
  const float coeff = coeff_p[0];
  const int n0 = blockIdx.x * 64;
  const int k0 = blockIdx.y * 64;
  const int t  = threadIdx.x;
#pragma unroll
  for (int i = 0; i < 16; ++i) {             // coalesced read of [K][N]
    int idx = t + i * 256;
    int r = idx >> 6, c = idx & 63;
    size_t g = (size_t)(k0 + r) * DIM_N + (n0 + c);
    float w = base[g] + coeff * (float)(2 * mask[g] - 1);
    tile[r * 65 + c] = w;
  }
  __syncthreads();
#pragma unroll
  for (int i = 0; i < 16; ++i) {             // coalesced write of WT[N][K]
    int idx = t + i * 256;
    int rr = idx >> 6, cc = idx & 63;        // rr: n-local, cc: k-local
    unsigned short h = f32_to_bf16(tile[cc * 65 + rr]);
    wt[(size_t)(n0 + rr) * DIM_K + (k0 + cc)] = h;
  }
}

// ---------------------------------------------------------------------------
// Prep 2: x f32 -> bf16 (one pass; halves GEMM A-panel re-read traffic).
// ---------------------------------------------------------------------------
__global__ __launch_bounds__(256) void bd_prep_x(const float* __restrict__ x,
                                                 unsigned short* __restrict__ xb) {
  size_t i = ((size_t)blockIdx.x * 256 + threadIdx.x) * 8;
  const float4* src = (const float4*)(x + i);
  float4 a = src[0], b = src[1];
  uint4 o;
  o.x = (unsigned)f32_to_bf16(a.x) | ((unsigned)f32_to_bf16(a.y) << 16);
  o.y = (unsigned)f32_to_bf16(a.z) | ((unsigned)f32_to_bf16(a.w) << 16);
  o.z = (unsigned)f32_to_bf16(b.x) | ((unsigned)f32_to_bf16(b.y) << 16);
  o.w = (unsigned)f32_to_bf16(b.z) | ((unsigned)f32_to_bf16(b.w) << 16);
  *(uint4*)(xb + i) = o;
}

// ---------------------------------------------------------------------------
// GEMM: C[M][N] = A[M][K](bf16) * WT[N][K]^T(bf16), f32 accumulate via
// v_wmma_f32_16x16x32_bf16.  Block = 256 thr = 8 waves, tile 128x128, KT=32,
// double-buffered LDS (40 KB), wave tile 32x64 -> 2x4 v8f accumulators.
// Staging: async global->LDS (no VGPR staging -> no spills), ASYNCcnt sync.
// ---------------------------------------------------------------------------
#define MT   128
#define NT   128
#define KT   32
#define LDSS 40   // 32 + 8 pad bf16 elems (80B rows, 16B aligned, bank-stagger)

__global__ __launch_bounds__(256, 1) void bd_gemm(const unsigned short* __restrict__ xb,
                                                  const unsigned short* __restrict__ wt,
                                                  float* __restrict__ out) {
  __shared__ __bf16 As[2][MT * LDSS];
  __shared__ __bf16 Bs[2][NT * LDSS];

  const int tid    = threadIdx.x;
  const int lane   = tid & 31;
  const int lrow   = lane & 15;
  const bool hih   = lane >= 16;
  const int wave   = tid >> 5;
  const int wm     = wave & 3;               // 4 waves along M (32 rows each)
  const int wn     = wave >> 2;              // 2 waves along N (64 cols each)
  const long m0    = (long)blockIdx.y * MT;
  const long n0    = (long)blockIdx.x * NT;

  const __bf16* A  = (const __bf16*)xb;      // [M][K]
  const __bf16* Bm = (const __bf16*)wt;      // [N][K]

  v8f acc[2][4] = {};

  // stage: 128 rows x 32 K x 2B = 8KB per matrix = 512 x 16B chunks; 2/thread
  auto issue_stage = [&](int k0, int buf) {
#pragma unroll
    for (int i = 0; i < 2; ++i) {
      int c = tid + i * 256;
      int row = c >> 2;
      int col = (c & 3) * 8;
#if HAVE_ASYNC_LDS
      __builtin_amdgcn_global_load_async_to_lds_b128(
          (v4i_g*)(void*)(A + (m0 + row) * DIM_K + k0 + col),
          (v4i_l*)(void*)&As[buf][row * LDSS + col],
          0, 0);
      __builtin_amdgcn_global_load_async_to_lds_b128(
          (v4i_g*)(void*)(Bm + (n0 + row) * DIM_K + k0 + col),
          (v4i_l*)(void*)&Bs[buf][row * LDSS + col],
          0, 0);
#else
      uint4 va = *(const uint4*)(A  + (m0 + row) * DIM_K + k0 + col);
      uint4 vb = *(const uint4*)(Bm + (n0 + row) * DIM_K + k0 + col);
      *(uint4*)&As[buf][row * LDSS + col] = va;
      *(uint4*)&Bs[buf][row * LDSS + col] = vb;
#endif
    }
  };
  auto wait_stage = [&]() {
#if HAVE_ASYNC_LDS
#if __has_builtin(__builtin_amdgcn_s_wait_asynccnt)
    __builtin_amdgcn_s_wait_asynccnt(0);
#else
    asm volatile("s_wait_asynccnt 0x0" ::: "memory");
#endif
#endif
  };
  auto compute = [&](int buf) {
    v16bf af[2], bfr[4];
    // A frag (16x32 bf16): lane<16 -> K[0..7],[16..23]; lane>=16 -> +8
#pragma unroll
    for (int mi = 0; mi < 2; ++mi) {
      const __bf16* p = &As[buf][(wm * 32 + mi * 16 + lrow) * LDSS + (hih ? 8 : 0)];
      v8bf lo = *(const v8bf*)p;
      v8bf hi = *(const v8bf*)(p + 16);
      af[mi] = __builtin_shufflevector(lo, hi, 0,1,2,3,4,5,6,7,8,9,10,11,12,13,14,15);
    }
    // B frag (32x16): lane n = lane&15; lane<16 -> K[0..7],[16..23]; lane>=16 -> +8
#pragma unroll
    for (int ni = 0; ni < 4; ++ni) {
      const __bf16* p = &Bs[buf][(wn * 64 + ni * 16 + lrow) * LDSS + (hih ? 8 : 0)];
      v8bf lo = *(const v8bf*)p;
      v8bf hi = *(const v8bf*)(p + 16);
      bfr[ni] = __builtin_shufflevector(lo, hi, 0,1,2,3,4,5,6,7,8,9,10,11,12,13,14,15);
    }
#pragma unroll
    for (int mi = 0; mi < 2; ++mi)
#pragma unroll
      for (int ni = 0; ni < 4; ++ni)
        acc[mi][ni] = __builtin_amdgcn_wmma_f32_16x16x32_bf16(
            false, af[mi], false, bfr[ni], (short)0, acc[mi][ni], false, false);
  };

  issue_stage(0, 0);
  wait_stage();
  __syncthreads();
  int buf = 0;
  for (int k0 = KT; k0 < DIM_K; k0 += KT) {
    issue_stage(k0, buf ^ 1);                // async fill of other buffer
    if (k0 + KT < DIM_K) {                   // L2 prefetch of stage after next
      __builtin_prefetch(A  + (m0 + (tid >> 1)) * DIM_K + k0 + KT, 0, 0);
      __builtin_prefetch(Bm + (n0 + (tid >> 1)) * DIM_K + k0 + KT, 0, 0);
    }
    compute(buf);                            // WMMA on current stage (overlaps)
    wait_stage();                            // our async loads done
    __syncthreads();                         // everyone's loads/reads done
    buf ^= 1;
  }
  compute(buf);

  // C/D layout: VGPR r -> row r (lanes 0-15) / row r+8 (lanes 16-31), col = lane&15
  const long rbase = m0 + wm * 32 + (hih ? 8 : 0);
  const long cbase = n0 + wn * 64 + lrow;
#pragma unroll
  for (int mi = 0; mi < 2; ++mi)
#pragma unroll
    for (int ni = 0; ni < 4; ++ni)
#pragma unroll
      for (int r = 0; r < 8; ++r)
        out[(rbase + mi * 16 + r) * DIM_N + cbase + ni * 16] = acc[mi][ni][r];
}

extern "C" void kernel_launch(void* const* d_in, const int* in_sizes, int n_in,
                              void* d_out, int out_size, void* d_ws, size_t ws_size,
                              hipStream_t stream) {
  const float* x     = (const float*)d_in[0];   // [4,2048,4096] f32
  const float* base  = (const float*)d_in[1];   // [4096,4096]  f32
  const float* coeff = (const float*)d_in[2];   // scalar f32
  const int*   mask  = (const int*)d_in[3];     // [4096,4096]  i32
  float* out = (float*)d_out;                   // [4,2048,4096] f32

  unsigned short* wt = (unsigned short*)d_ws;                  // WT[N][K] bf16: 32 MB
  unsigned short* xb = wt + (size_t)DIM_K * DIM_N;             // Xb[M][K] bf16: 64 MB

  bd_prep_w<<<dim3(DIM_N / 64, DIM_K / 64), 256, 0, stream>>>(base, mask, coeff, wt);
  bd_prep_x<<<(unsigned)(((size_t)DIM_M * DIM_K) / (256 * 8)), 256, 0, stream>>>(x, xb);
  bd_gemm<<<dim3(DIM_N / NT, DIM_M / MT), 256, 0, stream>>>(xb, wt, out);
}
